// VideoEncoderConvLSTM_69896297775371
// MI455X (gfx1250) — matrix operand order repaired and checked by reading, not compiled, not run on hardware
//
#include <hip/hip_runtime.h>
#include <hip/hip_bf16.h>

typedef __bf16 bf16;
typedef __attribute__((ext_vector_type(16))) __bf16 v16bf;
typedef __attribute__((ext_vector_type(8)))  __bf16 v8bf;
typedef __attribute__((ext_vector_type(8)))  float  v8f;

__device__ __forceinline__ v16bf cat16(v8bf lo, v8bf hi) {
    return __builtin_shufflevector(lo, hi,
        0,1,2,3,4,5,6,7,8,9,10,11,12,13,14,15);
}

// ---------------------------------------------------------------------------
// Weight repack: f32 [kh,kw,Ci,Co] -> bf16 WMMA-B-fragment-major layout.
// Packed element e = (((ntile*Q + q)*32 + lane)*16 + j), where
//   q = (kh*KW+kw)*(Ci/32) + cc,  klo = (lane<16)?0:16,
//   value = W[kk, cc*32 + klo + j, ntile*16 + (lane&15)]
// In the GEMM each lane reads its 16 contiguous bf16 (two b128s), and the
// per-tile pointer advances by a constant 512 elements per K-chunk.
// ---------------------------------------------------------------------------
__global__ void pack_weights(const float* __restrict__ src, bf16* __restrict__ dst,
                             int KK, int Ci, int Co, long total) {
    long e = (long)blockIdx.x * blockDim.x + threadIdx.x;
    if (e >= total) return;
    int Cc = Ci >> 5;
    int Q  = KK * Cc;
    int j    = (int)(e & 15);
    int lane = (int)((e >> 4) & 31);
    long r   = e >> 9;
    int q  = (int)(r % Q);
    int nt = (int)(r / Q);
    int cc = q % Cc;
    int kk = q / Cc;
    int klo = (lane < 16) ? 0 : 16;
    int ci  = cc * 32 + klo + j;
    int n   = nt * 16 + (lane & 15);
    dst[e] = (bf16)src[((long)kk * Ci + ci) * Co + n];
}

// ---------------------------------------------------------------------------
// Implicit-GEMM convolution with bf16 WMMA (f32 accumulate).
//   act   : bf16 activations [frames, H, W, Ci] (NHWC)
//   wpack : packed bf16 weights (see pack_weights)
//   cinit : optional f32 initial accumulator (recurrent: xz[t]); index is
//           frame*cFrameStride + cTOff + p*N + col. nullptr -> zero init.
//   outz  : f32 [M, N]
//   zpad  : >=1KB zero-filled guard region; out-of-image (SAME padding) lanes
//           redirect their A loads here, so loads are unconditional and the
//           inner loop has no EXEC manipulation or register zero-fills.
// One wave computes a 16(M) x 64(N) tile: per 32-wide K chunk it issues
// 4 v_wmma_f32_16x16x32_bf16 sharing one A fragment. All A/B addresses are
// strength-reduced induction pointers (+64B / +1024B per chunk).
// M mult of 16, N mult of 64, Ci mult of 32 (guaranteed by caller).
// ---------------------------------------------------------------------------
__global__ __launch_bounds__(256) void
wmma_conv_gemm(const bf16* __restrict__ act, const bf16* __restrict__ wpack,
               const float* __restrict__ cinit, float* __restrict__ outz,
               const bf16* __restrict__ zpad,
               int M, int N, int H, int W, int Ci, int KH, int KW,
               int stride, int pad, int Ho, int Wo,
               long cFrameStride, long cTOff) {
    int lane = threadIdx.x & 31;
    long tile = (long)blockIdx.x * 8 + (threadIdx.x >> 5);
    int nGroups = N >> 6;
    long nTilesTotal = (long)(M >> 4) * nGroups;
    if (tile >= nTilesTotal) return;              // wave-uniform exit

    int m0 = (int)(tile / nGroups) << 4;
    int n0 = (int)(tile % nGroups) << 6;
    int HoWo = Ho * Wo;
    int Cc = Ci >> 5;
    int Q  = KH * KW * Cc;
    int frame = m0 / HoWo;                        // tiles never cross frames
    int pbase = m0 - frame * HoWo;

    int p  = pbase + (lane & 15);                 // this lane's M row
    int ho = p / Wo, wo = p - ho * Wo;
    int klo8 = (lane & 16) ? 16 : 0;              // A K-half select
    int hi8  = (lane & 16) ? 8 : 0;               // C/D M-half select
    int ccol = lane & 15;

    v8f acc[4];
    if (cinit) {
        const float* cb = cinit + (long)frame * cFrameStride + cTOff;
#pragma unroll
        for (int j = 0; j < 4; ++j)
#pragma unroll
            for (int g = 0; g < 8; ++g)
                acc[j][g] = cb[(long)(pbase + g + hi8) * N + (n0 + j * 16 + ccol)];
    } else {
        v8f zero = {0.f, 0.f, 0.f, 0.f, 0.f, 0.f, 0.f, 0.f};
#pragma unroll
        for (int j = 0; j < 4; ++j) acc[j] = zero;
    }

    // Induction pointers: 4 B-fragment streams, +512 elements per K chunk.
    long jstride = (long)Q * 512;
    const bf16* bp0 = wpack + (long)(n0 >> 4) * jstride + lane * 16;
    const bf16* bp1 = bp0 + jstride;
    const bf16* bp2 = bp1 + jstride;
    const bf16* bp3 = bp2 + jstride;

    const bf16* fbase = act + (long)frame * H * W * Ci + klo8;
    const bf16* zbase = zpad + klo8;
    for (int kh = 0; kh < KH; ++kh) {
        int ih = ho * stride + kh - pad;
        bool vh = (unsigned)ih < (unsigned)H;
        const bf16* rbase = fbase + (long)ih * W * Ci;
        for (int kw = 0; kw < KW; ++kw) {
            int iw = wo * stride + kw - pad;
            bool valid = vh && ((unsigned)iw < (unsigned)W);
            // Per-lane address select: invalid (padding) lanes read zeros
            // from the guard page; keeps loads unconditional (full EXEC).
            const bf16* aptr = valid ? (rbase + (long)iw * Ci) : zbase;
            for (int cc = 0; cc < Cc; ++cc) {
                v8bf a0 = *(const v8bf*)aptr;
                v8bf a1 = *(const v8bf*)(aptr + 8);
                aptr += 32;
                v16bf Af = cat16(a0, a1);

                v8bf b0 = *(const v8bf*)bp0;
                v8bf b1 = *(const v8bf*)(bp0 + 8);
                acc[0] = __builtin_amdgcn_wmma_f32_16x16x32_bf16(
                    false, Af, false, cat16(b0, b1), (short)0, acc[0], false, false);
                b0 = *(const v8bf*)bp1;
                b1 = *(const v8bf*)(bp1 + 8);
                acc[1] = __builtin_amdgcn_wmma_f32_16x16x32_bf16(
                    false, Af, false, cat16(b0, b1), (short)0, acc[1], false, false);
                b0 = *(const v8bf*)bp2;
                b1 = *(const v8bf*)(bp2 + 8);
                acc[2] = __builtin_amdgcn_wmma_f32_16x16x32_bf16(
                    false, Af, false, cat16(b0, b1), (short)0, acc[2], false, false);
                b0 = *(const v8bf*)bp3;
                b1 = *(const v8bf*)(bp3 + 8);
                acc[3] = __builtin_amdgcn_wmma_f32_16x16x32_bf16(
                    false, Af, false, cat16(b0, b1), (short)0, acc[3], false, false);

                bp0 += 512; bp1 += 512; bp2 += 512; bp3 += 512;
            }
        }
    }

#pragma unroll
    for (int j = 0; j < 4; ++j)
#pragma unroll
        for (int g = 0; g < 8; ++g)
            outz[(long)(m0 + g + hi8) * N + (n0 + j * 16 + ccol)] = acc[j][g];
}

// ---------------------------------------------------------------------------
// Layer-1 input conv (Ci=3, K=108 -> not WMMA-friendly; ~1% of total FLOPs).
// x[b,t,h,w,c] = in[b,h,t,w,c]; stride 2, pad 2, kernel 6x6, Co=256.
// ---------------------------------------------------------------------------
__global__ void conv1_input(const float* __restrict__ in,
                            const float* __restrict__ k1,
                            float* __restrict__ xz) {
    long id = (long)blockIdx.x * blockDim.x + threadIdx.x;
    const long total = 128L * 1024 * 256;
    if (id >= total) return;
    int co = (int)(id & 255);
    int p  = (int)((id >> 8) & 1023);
    int bt = (int)(id >> 18);
    int t = bt & 15, b = bt >> 4;
    int ho = p >> 5, wo = p & 31;
    float acc = 0.f;
    for (int kh = 0; kh < 6; ++kh) {
        int ih = ho * 2 + kh - 2;
        if ((unsigned)ih >= 64u) continue;
        for (int kw = 0; kw < 6; ++kw) {
            int iw = wo * 2 + kw - 2;
            if ((unsigned)iw >= 64u) continue;
            const float* px = in + (((long)(b * 64 + ih) * 16 + t) * 64 + iw) * 3;
            const float* wk = k1 + (long)((kh * 6 + kw) * 3) * 256 + co;
            acc += px[0] * wk[0] + px[1] * wk[256] + px[2] * wk[512];
        }
    }
    xz[id] = acc;
}

// ---------------------------------------------------------------------------
// Fused LSTM gates + LayerNorm. One block per (b, ho, wo); F threads.
// Writes: f32 cell state, bf16 pre-LN h (recurrent operand),
//         bf16 LN(h) into the next layer's input sequence,
//         f32 LN(h) into d_out when t >= s.
// ---------------------------------------------------------------------------
__device__ __forceinline__ float hsig(float x) {
    return fminf(fmaxf(0.2f * x + 0.5f, 0.f), 1.f);
}

__global__ void gate_ln(const float* __restrict__ z, float* __restrict__ c,
                        bf16* __restrict__ hstate, bf16* __restrict__ hseq,
                        float* __restrict__ outFeat,
                        const float* __restrict__ gamma,
                        const float* __restrict__ beta,
                        int F, int HoWo, int T, int t, int s) {
    extern __shared__ float sm[];
    int ch  = threadIdx.x;
    int pos = blockIdx.x;                 // b*HoWo + p
    long zb = (long)pos * 4 * F;
    float zi = z[zb + ch];
    float zf = z[zb + F + ch];
    float zg = z[zb + 2 * F + ch];
    float zo = z[zb + 3 * F + ch];
    long cb = (long)pos * F + ch;
    float cn = hsig(zf) * c[cb] + hsig(zi) * tanhf(zg);
    float hn = hsig(zo) * tanhf(cn);
    c[cb] = cn;
    hstate[cb] = (bf16)hn;

    float* s1 = sm;
    float* s2 = sm + F;
    s1[ch] = hn;
    s2[ch] = hn * hn;
    __syncthreads();
    for (int off = F >> 1; off > 0; off >>= 1) {
        if (ch < off) { s1[ch] += s1[ch + off]; s2[ch] += s2[ch + off]; }
        __syncthreads();
    }
    float mean = s1[0] / F;
    float var  = s2[0] / F - mean * mean;
    float y = (hn - mean) * rsqrtf(var + 1e-3f) * gamma[ch] + beta[ch];

    int b = pos / HoWo, pp = pos - b * HoWo;
    hseq[(((long)b * T + t) * HoWo + pp) * F + ch] = (bf16)y;
    if (t >= s)
        outFeat[(((long)b * (T - s) + (t - s)) * HoWo + pp) * F + ch] = y;
}

// ---------------------------------------------------------------------------
// feats[0] = x[:, 7:] = transposed input slice [8,9,64,64,3]
// ---------------------------------------------------------------------------
__global__ void feat0_copy(const float* __restrict__ in, float* __restrict__ out) {
    long id = (long)blockIdx.x * blockDim.x + threadIdx.x;
    if (id >= 884736L) return;
    int cc = (int)(id % 3);
    long r = id / 3;
    int ww = (int)(r & 63); r >>= 6;
    int hh = (int)(r & 63); r >>= 6;
    int j  = (int)(r % 9);
    int b  = (int)(r / 9);
    out[id] = in[((((long)b * 64 + hh) * 16 + (7 + j)) * 64 + ww) * 3 + cc];
}

__global__ void fill_zero_f32(float* p, long n) {
    long i = (long)blockIdx.x * blockDim.x + threadIdx.x;
    if (i < n) p[i] = 0.f;
}
__global__ void fill_zero_b16(bf16* p, long n) {
    long i = (long)blockIdx.x * blockDim.x + threadIdx.x;
    if (i < n) p[i] = (bf16)0.f;
}

// ---------------------------------------------------------------------------
extern "C" void kernel_launch(void* const* d_in, const int* in_sizes, int n_in,
                              void* d_out, int out_size, void* d_ws, size_t ws_size,
                              hipStream_t stream) {
    (void)in_sizes; (void)n_in; (void)out_size; (void)ws_size;
    const float* x_in = (const float*)d_in[0];
    const float *Kw[4], *Rw[4], *Gm[4], *Bt[4];
    for (int l = 0; l < 4; ++l) {
        Kw[l] = (const float*)d_in[1 + 4 * l];
        Rw[l] = (const float*)d_in[2 + 4 * l];
        Gm[l] = (const float*)d_in[3 + 4 * l];
        Bt[l] = (const float*)d_in[4 + 4 * l];
    }
    float* out = (float*)d_out;

    // ---- carve workspace (~290 MB) ----
    size_t off = 0;
    char* base = (char*)d_ws;
    auto carve = [&](size_t bytes) -> char* {
        char* p = base + off;
        off += (bytes + 255) & ~(size_t)255;
        return p;
    };
    float* xz     = (float*)carve(33554432UL * 4); // max: L1 input conv output
    float* zbuf   = (float*)carve(2097152UL * 4);  // max per-step z (L1)
    float* cbuf   = (float*)carve(524288UL * 4);   // cell state (L1 max)
    bf16*  hstate = (bf16*) carve(524288UL * 2);   // pre-LN h, bf16
    bf16*  hseqA  = (bf16*) carve(8388608UL * 2);  // L1/L3 output sequence
    bf16*  hseqB  = (bf16*) carve(4194304UL * 2);  // L2/L4 output sequence
    bf16*  zpad   = (bf16*) carve(4096);           // zero guard page for padding

    const int KHs[4] = {6, 6, 5, 5};
    const int Cin[4] = {3, 64, 128, 256};
    const int Fc[4]  = {64, 128, 256, 512};
    const int Hin[4] = {64, 32, 16, 8};
    const int Hout[4]  = {32, 16, 8, 4};
    const int padIn[4] = {2, 2, 1, 1};   // SAME, stride 2
    const int padR[4]  = {2, 2, 2, 2};   // SAME, stride 1
    const size_t outOff[5] = {0, 884736, 5603328, 7962624, 9142272};
    const int B = 8, T = 16, S = 7;

    bf16* pk[4] = {nullptr, nullptr, nullptr, nullptr};
    bf16* pr[4];
    for (int l = 1; l < 4; ++l)
        pk[l] = (bf16*)carve((size_t)KHs[l] * KHs[l] * Cin[l] * 4 * Fc[l] * 2);
    for (int l = 0; l < 4; ++l)
        pr[l] = (bf16*)carve((size_t)KHs[l] * KHs[l] * Fc[l] * 4 * Fc[l] * 2);

    // Zero the padding guard page (must precede all GEMMs; stream-ordered).
    fill_zero_b16<<<8, 256, 0, stream>>>(zpad, 2048);

    // feats[0]
    feat0_copy<<<(884736 + 255) / 256, 256, 0, stream>>>(x_in, out);

    // Repack weights to bf16 WMMA-fragment layout (whole set ~107MB: resident
    // in the 192MB global L2 during the scan).
    for (int l = 1; l < 4; ++l) {
        long n = (long)KHs[l] * KHs[l] * Cin[l] * 4 * Fc[l];
        pack_weights<<<(unsigned)((n + 255) / 256), 256, 0, stream>>>(
            Kw[l], pk[l], KHs[l] * KHs[l], Cin[l], 4 * Fc[l], n);
    }
    for (int l = 0; l < 4; ++l) {
        long n = (long)KHs[l] * KHs[l] * Fc[l] * 4 * Fc[l];
        pack_weights<<<(unsigned)((n + 255) / 256), 256, 0, stream>>>(
            Rw[l], pr[l], KHs[l] * KHs[l], Fc[l], 4 * Fc[l], n);
    }

    bf16* seqIn = nullptr;
    for (int l = 0; l < 4; ++l) {
        int Ho = Hout[l], HoWo = Ho * Ho, N = 4 * Fc[l], F = Fc[l], KH = KHs[l];
        long Mfull = (long)B * T * HoWo;       // all frames (input conv)
        long Mstep = (long)B * HoWo;           // per timestep (recurrent)
        bf16* seqOut = (l & 1) ? hseqB : hseqA;

        // Hoisted input conv over all B*T frames.
        if (l == 0) {
            long n = Mfull * N;
            conv1_input<<<(unsigned)((n + 255) / 256), 256, 0, stream>>>(x_in, Kw[0], xz);
        } else {
            long tiles = (Mfull >> 4) * (N >> 6);
            wmma_conv_gemm<<<(unsigned)((tiles + 7) / 8), 256, 0, stream>>>(
                seqIn, pk[l], nullptr, xz, zpad, (int)Mfull, N,
                Hin[l], Hin[l], Cin[l], KH, KH, 2, padIn[l], Ho, Ho, 0L, 0L);
        }

        // h0 = c0 = 0
        long stateN = Mstep * F;
        fill_zero_f32<<<(unsigned)((stateN + 255) / 256), 256, 0, stream>>>(cbuf, stateN);
        fill_zero_b16<<<(unsigned)((stateN + 255) / 256), 256, 0, stream>>>(hstate, stateN);

        // Sequential scan: z = xz[t] + conv(h, R); gates; LN.
        for (int t = 0; t < T; ++t) {
            long tiles = (Mstep >> 4) * (N >> 6);
            wmma_conv_gemm<<<(unsigned)((tiles + 7) / 8), 256, 0, stream>>>(
                hstate, pr[l], xz, zbuf, zpad, (int)Mstep, N,
                Ho, Ho, F, KH, KH, 1, padR[l], Ho, Ho,
                (long)T * HoWo * N, (long)t * HoWo * N);
            gate_ln<<<(unsigned)Mstep, F, 2 * F * (int)sizeof(float), stream>>>(
                zbuf, cbuf, hstate, seqOut, out + outOff[l + 1],
                Gm[l], Bt[l], F, HoWo, T, t, S);
        }
        seqIn = seqOut;
    }
}
